// Qwen3Attention_56513179681209
// MI455X (gfx1250) — matrix operand order repaired and checked by reading, not compiled, not run on hardware
//
#include <hip/hip_runtime.h>
#include <hip/hip_bf16.h>

// ---------------------------------------------------------------------------
// Qwen3-style attention layer on MI455X (gfx1250, wave32, WMMA).
//   1) qkv = hidden @ w_qkv                      (f16 WMMA GEMM, f32 accum)
//   2) per-head RMSNorm + RoPE, repack to f16    (VALU)
//   3) causal GQA flash attention                (f16 WMMA, online softmax)
//   4) out = att @ w_o                           (f16 WMMA GEMM)
// ---------------------------------------------------------------------------

typedef __attribute__((ext_vector_type(16))) _Float16 v16h;
typedef __attribute__((ext_vector_type(8)))  _Float16 v8h;
typedef __attribute__((ext_vector_type(8)))  float    v8f;

__device__ __forceinline__ v16h join16(v8h lo, v8h hi) {
  return __builtin_shufflevector(lo, hi, 0,1,2,3,4,5,6,7,8,9,10,11,12,13,14,15);
}

__device__ __forceinline__ v8f wmma_f16(v16h a, v16h b, v8f c) {
  // D(f32,16x16) = A(f16,16x32) * B(f16,32x16) + C
  return __builtin_amdgcn_wmma_f32_16x16x32_f16(
      /*neg_a=*/false, a, /*neg_b=*/false, b,
      /*c_mod=*/(short)0, c, /*reuse_a=*/false, /*reuse_b=*/false);
}

// ---------------------------------------------------------------------------
// GEMM: C[M,N] = A[M,K] * B[K,N]   (fp32 in/out, f16 WMMA inside)
// Block: 256 threads (8 waves). Tile: 128(M) x 64(N), BK=32 (one WMMA K-step).
// Wave (wm,wn) owns a 32x32 sub-tile = 2x2 WMMA tiles.
// LDS rows are 40 halfs (80B) -> every fragment v8h load is 16B aligned.
// ---------------------------------------------------------------------------
__global__ __launch_bounds__(256)
void gemm_f32_f16wmma(const float* __restrict__ A, const float* __restrict__ B,
                      float* __restrict__ C, int M, int N, int K) {
  constexpr int BM = 128, BN = 64, BK = 32, LDA = 40, LDB = 40;
  __shared__ __align__(16) _Float16 As[BM * LDA];  // [row][k]
  __shared__ __align__(16) _Float16 Bs[BN * LDB];  // [n][k]  (transposed)

  const int tid  = threadIdx.x;
  const int lane = tid & 31;
  const int w    = tid >> 5;
  const int wm   = w >> 1;           // 0..3  -> M sub-tile
  const int wn   = w & 1;            // 0..1  -> N sub-tile
  const int hl   = lane >> 4;        // half-wave id
  const int l15  = lane & 15;
  const int bm   = blockIdx.y * BM;
  const int bn   = blockIdx.x * BN;

  v8f acc[2][2] = {};

  for (int k0 = 0; k0 < K; k0 += BK) {
    // ---- stage A tile (128x32) : 16 contiguous fp32 per thread -> f16 LDS
    {
      const int row = tid >> 1;
      const int kc  = (tid & 1) * 16;
      const float* ap = A + (size_t)(bm + row) * K + k0 + kc;
      if (k0 + BK < K) __builtin_prefetch(ap + BK, 0, 0);  // global_prefetch_b8
      _Float16* dst = &As[row * LDA + kc];
#pragma unroll
      for (int i = 0; i < 16; ++i) dst[i] = (_Float16)ap[i];
    }
    // ---- stage B tile (32x64), transposed into [n][k]
    {
      const int kr = tid >> 3;        // 0..31
      const int nc = (tid & 7) * 8;   // 0..56
      const float* bp = B + (size_t)(k0 + kr) * N + bn + nc;
      if (k0 + BK < K) __builtin_prefetch(bp + (size_t)BK * N, 0, 0);
#pragma unroll
      for (int i = 0; i < 8; ++i) Bs[(nc + i) * LDB + kr] = (_Float16)bp[i];
    }
    __syncthreads();

    // ---- fragments (contiguous k order, same in A and B) + 4 WMMAs
    v16h afr[2], bfr[2];
#pragma unroll
    for (int t = 0; t < 2; ++t) {
      // A: row = lane&15, K = hl*16 + e
      const _Float16* p = &As[(wm * 32 + t * 16 + l15) * LDA + hl * 16];
      afr[t] = join16(*(const v8h*)p, *(const v8h*)(p + 8));
    }
#pragma unroll
    for (int t = 0; t < 2; ++t) {
      // B: col = lane&15, K = hl*16 + e
      const _Float16* p = &Bs[(wn * 32 + t * 16 + l15) * LDB + hl * 16];
      bfr[t] = join16(*(const v8h*)p, *(const v8h*)(p + 8));
    }
#pragma unroll
    for (int i = 0; i < 2; ++i)
#pragma unroll
      for (int j = 0; j < 2; ++j)
        acc[i][j] = wmma_f16(afr[i], bfr[j], acc[i][j]);
    __syncthreads();
  }

  // ---- epilogue: C/D layout row = v + 8*hl, col = lane&15
#pragma unroll
  for (int i = 0; i < 2; ++i)
#pragma unroll
    for (int j = 0; j < 2; ++j) {
      const int rb = bm + wm * 32 + i * 16 + 8 * hl;
      const int cb = bn + wn * 32 + j * 16 + l15;
#pragma unroll
      for (int v = 0; v < 8; ++v)
        C[(size_t)(rb + v) * N + cb] = acc[i][j][v];
    }
}

// ---------------------------------------------------------------------------
// RMSNorm + RoPE + repack.  grid = (T, 32), block = 128 (one dim per thread).
//   bh 0..15  : Q head  -> Qf[h][t][d]  (f16, pre-scaled by HD^-0.5)
//   bh 16..23 : K head  -> Kf[h][t][d]  (f16)
//   bh 24..31 : V head  -> Vt[h][d][t]  (f16, transposed for PV B-fragments)
// ---------------------------------------------------------------------------
__global__ __launch_bounds__(128)
void prep_kernel(const int* __restrict__ positions, const float* __restrict__ qkv,
                 const float* __restrict__ qw, const float* __restrict__ kw,
                 _Float16* __restrict__ Qf, _Float16* __restrict__ Kf,
                 _Float16* __restrict__ Vt) {
  constexpr int T = 4096, HD = 128;
  const int t  = blockIdx.x;
  const int bh = blockIdx.y;
  const int d  = threadIdx.x;

  if (bh >= 24) {  // V: plain transpose copy
    const int h = bh - 24;
    const float v = qkv[(size_t)t * 4096 + 3072 + h * HD + d];
    Vt[((size_t)h * HD + d) * T + t] = (_Float16)v;
    return;
  }
  const bool isQ = bh < 16;
  const int h = isQ ? bh : bh - 16;
  const size_t base = (size_t)t * 4096 + (isQ ? h * HD : 2048 + h * HD);
  const float x = qkv[base + d];

  // RMS over HD=128 (4 waves: shuffle reduce + LDS combine)
  __shared__ float red[4];
  __shared__ float buf[128];
  float ss = x * x;
#pragma unroll
  for (int off = 16; off; off >>= 1) ss += __shfl_xor(ss, off, 32);
  if ((threadIdx.x & 31) == 0) red[threadIdx.x >> 5] = ss;
  __syncthreads();
  ss = red[0] + red[1] + red[2] + red[3];
  const float y = x * __frsqrt_rn(ss * (1.0f / 128.0f) + 1e-6f) * (isQ ? qw[d] : kw[d]);

  // RoPE (neox rotate-half): exchange with partner dim through LDS
  buf[d] = y;
  __syncthreads();
  const float y2 = buf[d < 64 ? d + 64 : d - 64];
  const int   fi = d < 64 ? d : d - 64;
  const float pos = (float)positions[t];
  // inv_freq = THETA^(-2*fi/HD) = exp(-fi * ln(1e6)/64)
  const float ang = pos * __expf(-(float)fi * (13.815510558f / 64.0f));
  const float c = __cosf(ang), s = __sinf(ang);
  const float out = (d < 64) ? (y * c - y2 * s) : (y * c + y2 * s);

  if (isQ) Qf[((size_t)h * T + t) * HD + d] = (_Float16)(out * 0.08838834764831845f);
  else     Kf[((size_t)h * T + t) * HD + d] = (_Float16)out;
}

// ---------------------------------------------------------------------------
// Causal GQA flash attention. grid = (T/128, NH), block = 256 (8 waves).
// Each wave owns 16 query rows; 32-key steps; WMMA for QK^T and PV.
// ---------------------------------------------------------------------------
__global__ __launch_bounds__(256)
void attn_kernel(const _Float16* __restrict__ Qf, const _Float16* __restrict__ Kf,
                 const _Float16* __restrict__ Vt, float* __restrict__ att) {
  constexpr int BQ = 128, BK = 32, HD = 128, T = 4096, LDP = 40, QS = 2048;
  __shared__ __align__(16) _Float16 Ps[8 * 16 * LDP];  // per-wave P bounce

  const int lane = threadIdx.x & 31;
  const int w    = threadIdx.x >> 5;
  const int hl   = lane >> 4;
  const int l15  = lane & 15;
  const int h    = blockIdx.y;
  const int kvh  = h >> 1;                 // G = NH/NKV = 2
  const int qb0  = blockIdx.x * BQ;
  const int qrow0 = qb0 + w * 16;
  _Float16* Pw = &Ps[w * 16 * LDP];

  // Q fragments held in registers for the whole key loop (4 K-steps of 32 dims)
  v16h qfr[4];
  {
    const _Float16* qrow = Qf + ((size_t)h * T + (qrow0 + l15)) * HD;
#pragma unroll
    for (int st = 0; st < 4; ++st) {
      const _Float16* p = qrow + st * 32 + hl * 16;
      qfr[st] = join16(*(const v8h*)p, *(const v8h*)(p + 8));
    }
  }

  v8f o[8] = {};
  float m[8], l[8];
#pragma unroll
  for (int v = 0; v < 8; ++v) { m[v] = -1e30f; l[v] = 0.0f; }

  const int kend = qb0 + BQ;  // causal horizon for this query block
  for (int kb = 0; kb < kend; kb += BK) {
    // ---- S = Q * K^T  (two 16-key tiles)
    v8f s0 = {}, s1 = {};
#pragma unroll
    for (int st = 0; st < 4; ++st) {
      const _Float16* k0p = Kf + ((size_t)kvh * T + (kb + l15)) * HD + st * 32 + hl * 16;
      const _Float16* k1p = k0p + 16 * HD;
      v16h b0 = join16(*(const v8h*)k0p, *(const v8h*)(k0p + 8));
      v16h b1 = join16(*(const v8h*)k1p, *(const v8h*)(k1p + 8));
      s0 = wmma_f16(qfr[st], b0, s0);
      s1 = wmma_f16(qfr[st], b1, s1);
    }
    // ---- causal mask on diagonal blocks
    if (kb + BK - 1 > qrow0) {
#pragma unroll
      for (int v = 0; v < 8; ++v) {
        const int row  = qrow0 + v + 8 * hl;
        const int key0 = kb + l15;
        if (key0 > row)      s0[v] = -1e30f;
        if (key0 + 16 > row) s1[v] = -1e30f;
      }
    }
    // ---- online softmax (row reductions stay inside a 16-lane half)
    float pscale[8];
#pragma unroll
    for (int v = 0; v < 8; ++v) {
      float t0 = fmaxf(s0[v], s1[v]);
#pragma unroll
      for (int off = 8; off; off >>= 1) t0 = fmaxf(t0, __shfl_xor(t0, off, 32));
      const float mn = fmaxf(m[v], t0);
      const float sc = __expf(m[v] - mn);
      const float p0 = __expf(s0[v] - mn);
      const float p1 = __expf(s1[v] - mn);
      float rs = p0 + p1;
#pragma unroll
      for (int off = 8; off; off >>= 1) rs += __shfl_xor(rs, off, 32);
      l[v] = l[v] * sc + rs;
      m[v] = mn;
      pscale[v] = sc;
      // P -> LDS in row-major so it can be re-read in A-fragment layout
      Pw[(v + 8 * hl) * LDP + l15]      = (_Float16)p0;
      Pw[(v + 8 * hl) * LDP + 16 + l15] = (_Float16)p1;
    }
#pragma unroll
    for (int c = 0; c < 8; ++c)
#pragma unroll
      for (int v = 0; v < 8; ++v) o[c][v] *= pscale[v];

    // ---- reload P as A fragment (16 rows x 32 keys), then O += P * V
    const _Float16* pp = &Pw[l15 * LDP + hl * 16];
    v16h pa = join16(*(const v8h*)pp, *(const v8h*)(pp + 8));
#pragma unroll
    for (int c = 0; c < 8; ++c) {
      // B: col = output dim (c*16 + l15), K = key; Vt rows are key-contiguous
      const _Float16* vp = Vt + ((size_t)kvh * HD + c * 16 + l15) * T + kb + hl * 16;
      v16h bv = join16(*(const v8h*)vp, *(const v8h*)(vp + 8));
      o[c] = wmma_f16(pa, bv, o[c]);
    }
  }

  // ---- normalize and write att[T, QS] (fp32, feeds the O-proj GEMM)
#pragma unroll
  for (int c = 0; c < 8; ++c)
#pragma unroll
    for (int v = 0; v < 8; ++v) {
      const int row = qrow0 + v + 8 * hl;
      const int col = h * HD + c * 16 + l15;
      att[(size_t)row * QS + col] = o[c][v] / l[v];
    }
}

// ---------------------------------------------------------------------------
extern "C" void kernel_launch(void* const* d_in, const int* in_sizes, int n_in,
                              void* d_out, int out_size, void* d_ws, size_t ws_size,
                              hipStream_t stream) {
  constexpr int T = 4096, H = 2048, QS = 2048, KVS = 1024, HD = 128;
  const int*   positions = (const int*)d_in[0];
  const float* hidden    = (const float*)d_in[1];
  const float* w_qkv     = (const float*)d_in[2];
  const float* w_o       = (const float*)d_in[3];
  const float* qw        = (const float*)d_in[4];
  const float* kw        = (const float*)d_in[5];
  float* out = (float*)d_out;

  // workspace layout (all buffers fully written before read)
  char* ws = (char*)d_ws;
  float*    qkv = (float*)ws;                                   // T*4096 f32 = 64MB
  float*    att = (float*)(ws + (size_t)T * 4096 * 4);          // T*QS  f32 = 32MB
  _Float16* Qf  = (_Float16*)(ws + (size_t)T * 4096 * 4 + (size_t)T * QS * 4);
  _Float16* Kf  = Qf + (size_t)16 * T * HD;                     // 16MB after Qf
  _Float16* Vt  = Kf + (size_t)8 * T * HD;                      // 8MB after Kf
  (void)in_sizes; (void)n_in; (void)out_size; (void)ws_size;

  // 1) qkv = hidden @ w_qkv        [4096 x 4096] = [4096 x 2048][2048 x 4096]
  gemm_f32_f16wmma<<<dim3((QS + 2 * KVS) / 64, T / 128), dim3(256), 0, stream>>>(
      hidden, w_qkv, qkv, T, QS + 2 * KVS, H);
  // 2) RMSNorm + RoPE + f16 repack
  prep_kernel<<<dim3(T, 32), dim3(128), 0, stream>>>(positions, qkv, qw, kw, Qf, Kf, Vt);
  // 3) causal GQA flash attention
  attn_kernel<<<dim3(T / 128, 16), dim3(256), 0, stream>>>(Qf, Kf, Vt, att);
  // 4) out = att @ w_o             [4096 x 2048] = [4096 x 2048][2048 x 2048]
  gemm_f32_f16wmma<<<dim3(H / 64, T / 128), dim3(256), 0, stream>>>(
      att, w_o, out, T, H, QS);
}
